// Att_60189671686752
// MI455X (gfx1250) — matrix operand order repaired and checked by reading, not compiled, not run on hardware
//
#include <hip/hip_runtime.h>

typedef __attribute__((ext_vector_type(16))) _Float16 v16h;
typedef __attribute__((ext_vector_type(8)))  float    v8f;

#define D 128
#define GN_EPS 1e-5f

__device__ __forceinline__ int lane_id() { return (int)(threadIdx.x & 31u); }

__device__ __forceinline__ v8f wmma_f16(v16h a, v16h b, v8f c) {
  return __builtin_amdgcn_wmma_f32_16x16x32_f16(false, a, false, b, (short)0, c, false, false);
}

// A fragment (16x32 f16) from row-major f16 LDS tile X[16][ldk].
// lane L: row = L%16 ; halves 0..7 -> K = kb*32 + (L<16?0:8) + j ; halves 8..15 -> +16
__device__ __forceinline__ v16h a_frag_lds(const _Float16* X, int ldk, int kb) {
  int lane = lane_id();
  const _Float16* p = X + (lane & 15) * ldk + kb * 32 + ((lane >> 4) << 3);
  v16h a;
#pragma unroll
  for (int j = 0; j < 8; ++j) { a[j] = p[j]; a[8 + j] = p[16 + j]; }
  return a;
}

// A fragment built from a global f32 row (converted to f16 on the fly)
__device__ __forceinline__ v16h a_frag_g32(const float* R, int kb) {
  int lane = lane_id();
  int off = kb * 32 + ((lane >> 4) << 3);
  v16h a;
#pragma unroll
  for (int j = 0; j < 8; ++j) {
    a[j]     = (_Float16)R[off + j];
    a[8 + j] = (_Float16)R[off + 16 + j];
  }
  return a;
}

// B fragment (32x16 f16) for D = A x B where B[k][n] = W[n][k], W row-major [N][K] in LDS.
// lane L: col n = nb*16 + L%16 ; halves 0..7 -> K = kb*32 + (L<16?0:8) + j ; halves 8..15 -> +16
__device__ __forceinline__ v16h b_frag_lds(const _Float16* W, int ldk, int nb, int kb) {
  int lane = lane_id();
  const _Float16* p = W + (nb * 16 + (lane & 15)) * ldk + kb * 32 + ((lane >> 4) << 3);
  v16h b;
#pragma unroll
  for (int j = 0; j < 8; ++j) { b[j] = p[j]; b[8 + j] = p[16 + j]; }
  return b;
}

// GroupNorm(1 group over 128 cols) + optional ReLU on a 16x128 tile held in
// WMMA f32 D layout: acc[nb][r] = elem(row = r + 8*(lane>=16), col = nb*16 + lane%16).
// Row stats via shfl_xor(1,2,4,8) which stays inside each 16-lane half (wave32).
__device__ __forceinline__ void gn_d_layout(v8f* acc, const float* w, const float* b, bool relu_out) {
  int lane = lane_id();
  float wc[8], bc[8];
#pragma unroll
  for (int nb = 0; nb < 8; ++nb) { int c = nb * 16 + (lane & 15); wc[nb] = w[c]; bc[nb] = b[c]; }
#pragma unroll
  for (int r = 0; r < 8; ++r) {
    float s = 0.f;
#pragma unroll
    for (int nb = 0; nb < 8; ++nb) s += acc[nb][r];
    s += __shfl_xor(s, 1, 32); s += __shfl_xor(s, 2, 32);
    s += __shfl_xor(s, 4, 32); s += __shfl_xor(s, 8, 32);
    float mu = s * (1.f / 128.f);
    float v = 0.f;
#pragma unroll
    for (int nb = 0; nb < 8; ++nb) { float t = acc[nb][r] - mu; v += t * t; }
    v += __shfl_xor(v, 1, 32); v += __shfl_xor(v, 2, 32);
    v += __shfl_xor(v, 4, 32); v += __shfl_xor(v, 8, 32);
    float rs = rsqrtf(v * (1.f / 128.f) + GN_EPS);
#pragma unroll
    for (int nb = 0; nb < 8; ++nb) {
      float y = (acc[nb][r] - mu) * rs * wc[nb] + bc[nb];
      acc[nb][r] = relu_out ? fmaxf(y, 0.f) : y;
    }
  }
}

// ---------------- Kernel 1: per-agent precompute -------------------------
// Qh = f16( relu(GN(agts @ Wq^T)) ) ; A0 = agts @ Wa^T  (scatter accumulator)
__global__ void __launch_bounds__(256) agent_pre_kernel(
    const float* __restrict__ agts, const float* __restrict__ Wq,
    const float* __restrict__ gqw, const float* __restrict__ gqb,
    const float* __restrict__ Wa, _Float16* __restrict__ Qh,
    float* __restrict__ A0, int n_agt) {
  __shared__ _Float16 sWq[D * D];
  __shared__ _Float16 sWa[D * D];
  for (int i = threadIdx.x; i < D * D; i += blockDim.x) {
    sWq[i] = (_Float16)Wq[i]; sWa[i] = (_Float16)Wa[i];
  }
  __syncthreads();

  int lane = lane_id();
  int wave = blockIdx.x * (blockDim.x >> 5) + (threadIdx.x >> 5);
  int a0 = wave * 16;
  if (a0 >= n_agt) return;

  int rowm = min(a0 + (lane & 15), n_agt - 1);
  const float* R = agts + (size_t)rowm * D;
  v16h af[4];
#pragma unroll
  for (int kb = 0; kb < 4; ++kb) af[kb] = a_frag_g32(R, kb);

  v8f acc[8] = {};
#pragma unroll
  for (int kb = 0; kb < 4; ++kb)
#pragma unroll
    for (int nb = 0; nb < 8; ++nb)
      acc[nb] = wmma_f16(af[kb], b_frag_lds(sWq, D, nb, kb), acc[nb]);
  gn_d_layout(acc, gqw, gqb, true);
#pragma unroll
  for (int r = 0; r < 8; ++r) {
    int row = a0 + r + ((lane >> 4) << 3);
    if (row < n_agt) {
#pragma unroll
      for (int nb = 0; nb < 8; ++nb)
        Qh[(size_t)row * D + nb * 16 + (lane & 15)] = (_Float16)acc[nb][r];
    }
  }

  v8f acc2[8] = {};
#pragma unroll
  for (int kb = 0; kb < 4; ++kb)
#pragma unroll
    for (int nb = 0; nb < 8; ++nb)
      acc2[nb] = wmma_f16(af[kb], b_frag_lds(sWa, D, nb, kb), acc2[nb]);
#pragma unroll
  for (int r = 0; r < 8; ++r) {
    int row = a0 + r + ((lane >> 4) << 3);
    if (row < n_agt) {
#pragma unroll
      for (int nb = 0; nb < 8; ++nb)
        A0[(size_t)row * D + nb * 16 + (lane & 15)] = acc2[nb][r];
    }
  }
}

// ---------------- Kernel 2: fused edge pipeline --------------------------
// 8 waves/WG: weights (160KB f16) + 8x 16x384 f16 stage = ~262KB LDS (<320KB/WGP)
__global__ void __launch_bounds__(256) edge_kernel(
    const float* __restrict__ agt_ctrs, const float* __restrict__ ctx_ctrs,
    const int* __restrict__ hi, const int* __restrict__ wi,
    const float* __restrict__ Wd1, const float* __restrict__ bd1,
    const float* __restrict__ Wd2, const float* __restrict__ gd2w, const float* __restrict__ gd2b,
    const float* __restrict__ Wc1, const float* __restrict__ gc1w, const float* __restrict__ gc1b,
    const float* __restrict__ Wc2, const float* __restrict__ ctx,
    const _Float16* __restrict__ Qh, float* __restrict__ A0, int n_edge) {
  __shared__ _Float16 sWd2[D * D];
  __shared__ _Float16 sWc1[D * 3 * D];
  __shared__ _Float16 sWc2[D * D];
  __shared__ __align__(16) _Float16 sStage[8 * 16 * 3 * D];  // per-wave 16x384 f16 concat buffer
  __shared__ float sWd1[D * 2];
  __shared__ float sbd1[D];

  for (int i = threadIdx.x; i < D * D; i += blockDim.x) {
    sWd2[i] = (_Float16)Wd2[i]; sWc2[i] = (_Float16)Wc2[i];
  }
  for (int i = threadIdx.x; i < D * 3 * D; i += blockDim.x) sWc1[i] = (_Float16)Wc1[i];
  for (int i = threadIdx.x; i < D * 2; i += blockDim.x) sWd1[i] = Wd1[i];
  for (int i = threadIdx.x; i < D; i += blockDim.x) sbd1[i] = bd1[i];
  __syncthreads();

  int lane = lane_id();
  int wv = threadIdx.x >> 5;
  _Float16* stage = sStage + wv * 16 * (3 * D);
  int half = lane >> 4;       // 0 or 1
  int m = lane & 15;          // my edge row within tile
  int nwaves = gridDim.x * (blockDim.x >> 5);
  int wave_g = blockIdx.x * (blockDim.x >> 5) + wv;
  int ntiles = (n_edge + 15) >> 4;

  for (int t = wave_g; t < ntiles; t += nwaves) {
    int e0 = t * 16;
    int em = min(e0 + m, n_edge - 1);
    int ai = hi[em];
    int ci = wi[em];
    float dx = agt_ctrs[ai * 2]     - ctx_ctrs[ci * 2];
    float dy = agt_ctrs[ai * 2 + 1] - ctx_ctrs[ci * 2 + 1];

    // gather Q (f16) into cols [128,256) and ctx (f32->f16) into cols [256,384)
    {
      const uint4* qsrc = (const uint4*)(Qh + (size_t)ai * D) + half * 8;
      uint4* qdst = (uint4*)(stage + m * (3 * D) + D) + half * 8;
#pragma unroll
      for (int j = 0; j < 8; ++j) qdst[j] = qsrc[j];
      const float4* csrc = (const float4*)(ctx + (size_t)ci * D) + half * 16;
      _Float16* cdst = stage + m * (3 * D) + 2 * D + half * 64;
#pragma unroll
      for (int j = 0; j < 16; ++j) {
        float4 c4 = csrc[j];
        cdst[j * 4 + 0] = (_Float16)c4.x; cdst[j * 4 + 1] = (_Float16)c4.y;
        cdst[j * 4 + 2] = (_Float16)c4.z; cdst[j * 4 + 3] = (_Float16)c4.w;
      }
    }

    // GEMM1: h2 = GN-relu( relu(d2 @ Wd1^T + bd1) @ Wd2^T ) ; h1 built in A-frag layout
    v8f acc[8] = {};
#pragma unroll
    for (int kb = 0; kb < 4; ++kb) {
      int kbase = kb * 32 + (half << 3);
      v16h a;
#pragma unroll
      for (int j = 0; j < 8; ++j) {
        int k0 = kbase + j, k1 = kbase + 16 + j;
        float h0 = fmaxf(sWd1[k0 * 2] * dx + sWd1[k0 * 2 + 1] * dy + sbd1[k0], 0.f);
        float h1 = fmaxf(sWd1[k1 * 2] * dx + sWd1[k1 * 2 + 1] * dy + sbd1[k1], 0.f);
        a[j] = (_Float16)h0; a[8 + j] = (_Float16)h1;
      }
#pragma unroll
      for (int nb = 0; nb < 8; ++nb)
        acc[nb] = wmma_f16(a, b_frag_lds(sWd2, D, nb, kb), acc[nb]);
    }
    gn_d_layout(acc, gd2w, gd2b, true);
#pragma unroll
    for (int r = 0; r < 8; ++r) {
      int row = r + (half << 3);
#pragma unroll
      for (int nb = 0; nb < 8; ++nb)
        stage[row * (3 * D) + nb * 16 + m] = (_Float16)acc[nb][r];
    }

    // GEMM2: 16x128 = concat(16x384) @ Wc1^T, then GN+relu
    v8f acc2[8] = {};
#pragma unroll
    for (int kb = 0; kb < 12; ++kb) {
      v16h a = a_frag_lds(stage, 3 * D, kb);
#pragma unroll
      for (int nb = 0; nb < 8; ++nb)
        acc2[nb] = wmma_f16(a, b_frag_lds(sWc1, 3 * D, nb, kb), acc2[nb]);
    }
    gn_d_layout(acc2, gc1w, gc1b, true);
#pragma unroll
    for (int r = 0; r < 8; ++r) {
      int row = r + (half << 3);
#pragma unroll
      for (int nb = 0; nb < 8; ++nb)
        stage[row * (3 * D) + nb * 16 + m] = (_Float16)acc2[nb][r];  // reuse cols 0..127
    }

    // GEMM3: c = h3 @ Wc2^T, scatter-add into A0[hi]
    v8f acc3[8] = {};
#pragma unroll
    for (int kb = 0; kb < 4; ++kb) {
      v16h a = a_frag_lds(stage, 3 * D, kb);
#pragma unroll
      for (int nb = 0; nb < 8; ++nb)
        acc3[nb] = wmma_f16(a, b_frag_lds(sWc2, D, nb, kb), acc3[nb]);
    }
#pragma unroll
    for (int r = 0; r < 8; ++r) {
      int rowm = r + (half << 3);
      int e = e0 + rowm;
      if (e < n_edge) {
        int ag = __shfl(ai, rowm, 32);
#pragma unroll
        for (int nb = 0; nb < 8; ++nb)
          unsafeAtomicAdd(&A0[(size_t)ag * D + nb * 16 + m], acc3[nb][r]);
      }
    }
  }
}

// ---------------- Kernel 3: per-agent post -------------------------------
// out = relu( GN( relu(GN(A0)) @ Wl^T ) + agts )
__global__ void __launch_bounds__(256) agent_post_kernel(
    const float* __restrict__ A0, const float* __restrict__ gnw, const float* __restrict__ gnb,
    const float* __restrict__ Wl, const float* __restrict__ glw, const float* __restrict__ glb,
    const float* __restrict__ agts, float* __restrict__ out, int n_agt) {
  __shared__ _Float16 sWl[D * D];
  for (int i = threadIdx.x; i < D * D; i += blockDim.x) sWl[i] = (_Float16)Wl[i];
  __syncthreads();

  int lane = lane_id();
  int wave = blockIdx.x * (blockDim.x >> 5) + (threadIdx.x >> 5);
  int a0 = wave * 16;
  if (a0 >= n_agt) return;

  int rowm = min(a0 + (lane & 15), n_agt - 1);
  const float* R = A0 + (size_t)rowm * D;
  int off = (lane >> 4) << 3;

  // lanes L and L^16 jointly hold row L%16 (disjoint 64-element halves)
  float x[64];
  float s = 0.f;
#pragma unroll
  for (int kb = 0; kb < 4; ++kb) {
    int kbase = kb * 32 + off;
#pragma unroll
    for (int j = 0; j < 8; ++j) {
      float v0 = R[kbase + j], v1 = R[kbase + 16 + j];
      x[kb * 16 + j] = v0; x[kb * 16 + 8 + j] = v1;
      s += v0 + v1;
    }
  }
  s += __shfl_xor(s, 16, 32);
  float mu = s * (1.f / 128.f);
  float v = 0.f;
#pragma unroll
  for (int i = 0; i < 64; ++i) { float t = x[i] - mu; v += t * t; }
  v += __shfl_xor(v, 16, 32);
  float rs = rsqrtf(v * (1.f / 128.f) + GN_EPS);

  v16h af[4];
#pragma unroll
  for (int kb = 0; kb < 4; ++kb) {
    int kbase = kb * 32 + off;
#pragma unroll
    for (int j = 0; j < 8; ++j) {
      int k0 = kbase + j, k1 = kbase + 16 + j;
      float y0 = fmaxf((x[kb * 16 + j]     - mu) * rs * gnw[k0] + gnb[k0], 0.f);
      float y1 = fmaxf((x[kb * 16 + 8 + j] - mu) * rs * gnw[k1] + gnb[k1], 0.f);
      af[kb][j] = (_Float16)y0; af[kb][8 + j] = (_Float16)y1;
    }
  }

  v8f acc[8] = {};
#pragma unroll
  for (int kb = 0; kb < 4; ++kb)
#pragma unroll
    for (int nb = 0; nb < 8; ++nb)
      acc[nb] = wmma_f16(af[kb], b_frag_lds(sWl, D, nb, kb), acc[nb]);
  gn_d_layout(acc, glw, glb, false);

#pragma unroll
  for (int r = 0; r < 8; ++r) {
    int row = a0 + r + ((lane >> 4) << 3);
    if (row < n_agt) {
#pragma unroll
      for (int nb = 0; nb < 8; ++nb) {
        int c = nb * 16 + (lane & 15);
        out[(size_t)row * D + c] = fmaxf(acc[nb][r] + agts[(size_t)row * D + c], 0.f);
      }
    }
  }
}

extern "C" void kernel_launch(void* const* d_in, const int* in_sizes, int n_in,
                              void* d_out, int out_size, void* d_ws, size_t ws_size,
                              hipStream_t stream) {
  const float* agts     = (const float*)d_in[0];
  const float* ctx      = (const float*)d_in[1];
  const float* agt_ctrs = (const float*)d_in[2];
  const float* ctx_ctrs = (const float*)d_in[3];
  const int*   hi       = (const int*)d_in[4];
  const int*   wi       = (const int*)d_in[5];
  const float* Wd1      = (const float*)d_in[6];
  const float* bd1      = (const float*)d_in[7];
  const float* Wd2      = (const float*)d_in[8];
  const float* gd2w     = (const float*)d_in[9];
  const float* gd2b     = (const float*)d_in[10];
  const float* Wq       = (const float*)d_in[11];
  const float* gqw      = (const float*)d_in[12];
  const float* gqb      = (const float*)d_in[13];
  const float* Wc1      = (const float*)d_in[14];
  const float* gc1w     = (const float*)d_in[15];
  const float* gc1b     = (const float*)d_in[16];
  const float* Wc2      = (const float*)d_in[17];
  const float* Wa       = (const float*)d_in[18];
  const float* gnw      = (const float*)d_in[19];
  const float* gnb      = (const float*)d_in[20];
  const float* Wl       = (const float*)d_in[21];
  const float* glw      = (const float*)d_in[22];
  const float* glb      = (const float*)d_in[23];

  int n_agt  = in_sizes[0] / D;
  int n_edge = in_sizes[4];

  _Float16* Qh = (_Float16*)d_ws;
  float* A0 = (float*)((char*)d_ws + (size_t)n_agt * D * sizeof(_Float16));
  float* outp = (float*)d_out;

  int waves_a  = (n_agt + 15) / 16;
  int blocks_a = (waves_a + 7) / 8;

  agent_pre_kernel<<<blocks_a, 256, 0, stream>>>(agts, Wq, gqw, gqb, Wa, Qh, A0, n_agt);
  edge_kernel<<<512, 256, 0, stream>>>(agt_ctrs, ctx_ctrs, hi, wi, Wd1, bd1,
                                       Wd2, gd2w, gd2b, Wc1, gc1w, gc1b, Wc2,
                                       ctx, Qh, A0, n_edge);
  agent_post_kernel<<<blocks_a, 256, 0, stream>>>(A0, gnw, gnb, Wl, glw, glb, agts, outp, n_agt);
}
